// GraphAttentionNetwork_12532714570466
// MI455X (gfx1250) — matrix operand order repaired and checked by reading, not compiled, not run on hardware
//
#include <hip/hip_runtime.h>
#include <hip/hip_bf16.h>

// ---------------------------------------------------------------------------
// Types / WMMA helpers (CDNA5 gfx1250, wave32)
// ---------------------------------------------------------------------------
typedef __bf16 v16bf  __attribute__((ext_vector_type(16)));
typedef __bf16 bf16x2 __attribute__((ext_vector_type(2)));
typedef float  v8f    __attribute__((ext_vector_type(8)));

union FragU { v16bf v; uint4 q[2]; };

__device__ __forceinline__ v8f zero8() {
  v8f r;
#pragma unroll
  for (int i = 0; i < 8; ++i) r[i] = 0.0f;
  return r;
}

// A-fragment (16x32 bf16): lane half 0 -> K 0..7 & 16..23, half 1 -> K 8..15 & 24..31
// rowp points at the start of this lane's row (row = lane & 15) in LDS.
__device__ __forceinline__ v16bf ld_afrag(const __bf16* rowp, int kbase, int lane) {
  const __bf16* p = rowp + kbase + ((lane >> 4) << 3);
  FragU u;
  u.q[0] = *(const uint4*)(p);
  u.q[1] = *(const uint4*)(p + 16);
  return u.v;
}

// B-fragment (32x16 bf16) from a column-major (Bt[n][k]) LDS tile:
// lane half g holds K = kbase + 8g .. +7 & kbase + 16 + 8g .. +7 for its column (col = lane & 15).
__device__ __forceinline__ v16bf ld_bfrag(const __bf16* colp, int kbase, int lane) {
  const __bf16* p = colp + kbase + ((lane >> 4) << 3);
  FragU u;
  u.q[0] = *(const uint4*)(p);
  u.q[1] = *(const uint4*)(p + 16);
  return u.v;
}

__device__ __forceinline__ v8f wmma_bf16(v16bf a, v16bf b, v8f c) {
  return __builtin_amdgcn_wmma_f32_16x16x32_bf16(false, a, false, b, (short)0, c,
                                                 false, false);
}

// CDNA5 async global->LDS copy (ASYNCcnt-tracked, no VGPR data path).
// LDS operand is the byte offset within the workgroup's LDS allocation
// (low 32 bits of the flat shared-aperture address, per ISA sec. 10.2).
__device__ __forceinline__ void async_copy_b128(const void* gptr, void* lptr) {
  unsigned lds = (unsigned)(uintptr_t)lptr;
  unsigned long long ga = (unsigned long long)(uintptr_t)gptr;
  asm volatile("global_load_async_to_lds_b128 %0, %1, off"
               :: "v"(lds), "v"(ga) : "memory");
}
__device__ __forceinline__ void wait_async0() {
  asm volatile("s_wait_asynccnt 0" ::: "memory");
}

// ---------------------------------------------------------------------------
// Problem constants
// ---------------------------------------------------------------------------
constexpr int Bc   = 8;
constexpr int Sc   = 512;
constexpr int INc  = 512;
constexpr int Hc   = 768;
constexpr int NHc  = 12;
constexpr int DHc  = 64;
constexpr int INTc = 3072;
constexpr int Lc   = 6;
constexpr int Mr   = Bc * Sc;       // 4096 rows of activations

// ---------------------------------------------------------------------------
// Kernel 1: xb = bf16(x + pos_table)
// ---------------------------------------------------------------------------
__global__ __launch_bounds__(256) void xpos_kernel(const float* __restrict__ x,
                                                   const float* __restrict__ pos,
                                                   __bf16* __restrict__ xb) {
  size_t i = (size_t)blockIdx.x * 256 + threadIdx.x;   // < Mr*INc
  int col = (int)(i % INc);
  int s   = (int)((i / INc) % Sc);
  xb[i] = (__bf16)(x[i] + pos[(size_t)s * INc + col]);
}

// ---------------------------------------------------------------------------
// Kernel 2: generic bf16 WMMA GEMM, C[M,N] = A[M,K] @ Bw[K,N] (+epilogue)
// BM=BN=128, BK=64, 256 threads = 8 waves (4x2), wave tile 32x64.
// ---------------------------------------------------------------------------
template <bool BIAS, bool RES, bool GELU, bool OUTF, bool OUTB>
__global__ __launch_bounds__(256) void gemm_kernel(
    const __bf16* __restrict__ A, const float* __restrict__ Bw,
    int M, int N, int K,
    const float* __restrict__ bias, const float* __restrict__ res,
    float* __restrict__ outF, __bf16* __restrict__ outB) {
  __shared__ __attribute__((aligned(16))) __bf16 sA[128 * 64];   // row-major, ld 64
  __shared__ __attribute__((aligned(16))) __bf16 sBt[128 * 64];  // Bt[n][k], ld 64

  const int tid  = threadIdx.x;
  const int lane = tid & 31;
  const int wave = tid >> 5;
  const int wm   = wave >> 1;      // 0..3  (32-row group)
  const int wn   = wave & 1;       // 0..1  (64-col group)
  const int bm   = blockIdx.y * 128;
  const int bn   = blockIdx.x * 128;
  const int g    = lane >> 4;
  const int n    = lane & 15;

  v8f acc[2][4];
#pragma unroll
  for (int a = 0; a < 2; ++a)
#pragma unroll
    for (int t = 0; t < 4; ++t) acc[a][t] = zero8();

  for (int k0 = 0; k0 < K; k0 += 64) {
    __syncthreads();
    {  // stage A tile: async global->LDS (bf16, no conversion needed)
      int r = tid >> 1, c = (tid & 1) * 32;
      const __bf16* gp = A + (size_t)(bm + r) * K + k0 + c;
      __bf16* lp = &sA[r * 64 + c];
#pragma unroll
      for (int i = 0; i < 4; ++i) async_copy_b128(gp + i * 8, lp + i * 8);
    }
    {  // stage B tile: fp32 global -> packed bf16x2, transposed in LDS
      int kr2 = (tid >> 3) * 2;        // two adjacent k rows
      int nc  = (tid & 7) * 16;        // 16 columns
      const float* gp0 = Bw + (size_t)(k0 + kr2) * N + bn + nc;
      const float* gp1 = gp0 + N;
#pragma unroll
      for (int i = 0; i < 16; ++i) {
        bf16x2 pk;
        pk[0] = (__bf16)gp0[i];
        pk[1] = (__bf16)gp1[i];
        *(bf16x2*)&sBt[(nc + i) * 64 + kr2] = pk;
      }
    }
    wait_async0();
    __syncthreads();

#pragma unroll
    for (int kk = 0; kk < 64; kk += 32) {
      v16bf af[2], bf[4];
#pragma unroll
      for (int a = 0; a < 2; ++a)
        af[a] = ld_afrag(sA + (wm * 32 + a * 16 + n) * 64, kk, lane);
#pragma unroll
      for (int t = 0; t < 4; ++t)
        bf[t] = ld_bfrag(sBt + (wn * 64 + t * 16 + n) * 64, kk, lane);
#pragma unroll
      for (int a = 0; a < 2; ++a)
#pragma unroll
        for (int t = 0; t < 4; ++t) acc[a][t] = wmma_bf16(af[a], bf[t], acc[a][t]);
    }
  }

  // Epilogue. C layout: VGPR v of lane (g,n) holds (row = v + 8g, col = n).
#pragma unroll
  for (int a = 0; a < 2; ++a) {
#pragma unroll
    for (int t = 0; t < 4; ++t) {
      const int col   = bn + wn * 64 + t * 16 + n;
      const int rbase = bm + wm * 32 + a * 16 + 8 * g;
      const float bcol = BIAS ? bias[col] : 0.0f;
#pragma unroll
      for (int v = 0; v < 8; ++v) {
        const size_t idx = (size_t)(rbase + v) * N + col;
        float xv = acc[a][t][v] + bcol;
        if (RES) xv += res[idx];
        if (GELU) xv = 0.5f * xv * (1.0f + erff(xv * 0.70710678118654752f));
        if (OUTF) outF[idx] = xv;
        if (OUTB) outB[idx] = (__bf16)xv;
      }
    }
  }
}

// ---------------------------------------------------------------------------
// Kernel 3: fused attention. One block = (b, head, 128 q rows); 8 waves,
// each wave owns 16 q rows. K/V streamed in 32-row chunks, online softmax.
// ---------------------------------------------------------------------------
__global__ __launch_bounds__(256) void attn_kernel(
    const __bf16* __restrict__ Q, const __bf16* __restrict__ Km,
    const __bf16* __restrict__ V, const float* __restrict__ graph,
    __bf16* __restrict__ ctx) {
  __shared__ __attribute__((aligned(16))) __bf16 sQ[128 * 64];  // [q][d], ld 64
  __shared__ __attribute__((aligned(16))) __bf16 sK[32 * 64];   // [kcol][d] == Bt
  __shared__ __attribute__((aligned(16))) __bf16 sVt[64 * 32];  // [d][s]   == Bt
  __shared__ __attribute__((aligned(16))) __bf16 sP[8][16 * 32];

  const int tid  = threadIdx.x;
  const int lane = tid & 31;
  const int wave = tid >> 5;
  const int qblk = blockIdx.x & 3;               // S/128 = 4
  const int head = (blockIdx.x >> 2) % NHc;
  const int b    = blockIdx.x / (4 * NHc);
  const int hoff = head * DHc;
  const size_t rowbase = (size_t)b * Sc;
  const int q0 = qblk * 128;
  const int g = lane >> 4, n = lane & 15;

  {  // stage Q tile once (async; covered by the in-loop asynccnt fence)
    int r = tid >> 1, c = (tid & 1) * 32;
    const __bf16* gp = Q + (rowbase + q0 + r) * Hc + hoff + c;
    __bf16* lp = &sQ[r * 64 + c];
#pragma unroll
    for (int i = 0; i < 4; ++i) async_copy_b128(gp + i * 8, lp + i * 8);
  }

  float mrow[8], lsum[8];
  v8f o[4];
#pragma unroll
  for (int v = 0; v < 8; ++v) { mrow[v] = -1e30f; lsum[v] = 0.0f; }
#pragma unroll
  for (int t = 0; t < 4; ++t) o[t] = zero8();

  for (int sc = 0; sc < Sc; sc += 32) {
    __syncthreads();
    {  // K chunk: row-major head slice == Bt layout for Q@K^T (async copy)
      int r = tid >> 3, c = (tid & 7) * 8;
      async_copy_b128(Km + (rowbase + sc + r) * Hc + hoff + c, &sK[r * 64 + c]);
    }
    {  // V chunk transposed into sVt[d][s]: two s-rows -> packed b32 stores
      int s2 = (tid >> 4) * 2;        // two adjacent s rows (0..30)
      int c4 = (tid & 15) * 4;        // 4 d columns
      const __bf16* gp0 = V + (rowbase + sc + s2) * Hc + hoff + c4;
      const __bf16* gp1 = gp0 + Hc;
#pragma unroll
      for (int i = 0; i < 4; ++i) {
        bf16x2 pk;
        pk[0] = gp0[i];
        pk[1] = gp1[i];
        *(bf16x2*)&sVt[(c4 + i) * 32 + s2] = pk;
      }
    }
    wait_async0();
    __syncthreads();

    // scores: two 16x16 tiles over k-columns [sc+0..15], [sc+16..31]
    v16bf aq0 = ld_afrag(sQ + (wave * 16 + n) * 64, 0, lane);
    v16bf aq1 = ld_afrag(sQ + (wave * 16 + n) * 64, 32, lane);
    float p[2][8];
#pragma unroll
    for (int t = 0; t < 2; ++t) {
      v8f s = zero8();
      s = wmma_bf16(aq0, ld_bfrag(sK + (t * 16 + n) * 64, 0, lane), s);
      s = wmma_bf16(aq1, ld_bfrag(sK + (t * 16 + n) * 64, 32, lane), s);
      const float* gr =
          graph + ((size_t)b * Sc + q0 + wave * 16 + 8 * g) * Sc + sc + t * 16 + n;
#pragma unroll
      for (int v = 0; v < 8; ++v)
        p[t][v] = s[v] * 0.125f + (1.0f - gr[(size_t)v * Sc]) * -10000.0f;
    }

    // online softmax (row stats replicated across the 16 lanes of each half)
#pragma unroll
    for (int v = 0; v < 8; ++v) {
      float rm = fmaxf(p[0][v], p[1][v]);
#pragma unroll
      for (int off = 8; off > 0; off >>= 1) rm = fmaxf(rm, __shfl_xor(rm, off, 32));
      float nm = fmaxf(mrow[v], rm);
      float f  = __expf(mrow[v] - nm);
      float p0 = __expf(p[0][v] - nm);
      float p1 = __expf(p[1][v] - nm);
      p[0][v] = p0; p[1][v] = p1;
      float rs = p0 + p1;
#pragma unroll
      for (int off = 8; off > 0; off >>= 1) rs += __shfl_xor(rs, off, 32);
      lsum[v] = lsum[v] * f + rs;
      mrow[v] = nm;
#pragma unroll
      for (int t = 0; t < 4; ++t) o[t][v] *= f;
    }

    // re-layout P (C layout -> A fragment) through wave-private LDS
    __bf16* pw = &sP[wave][0];
#pragma unroll
    for (int v = 0; v < 8; ++v) {
      pw[(v + 8 * g) * 32 + n]      = (__bf16)p[0][v];
      pw[(v + 8 * g) * 32 + 16 + n] = (__bf16)p[1][v];
    }
    asm volatile("s_wait_dscnt 0" ::: "memory");
    v16bf ap = ld_afrag(pw + n * 32, 0, lane);

#pragma unroll
    for (int dt = 0; dt < 4; ++dt)
      o[dt] = wmma_bf16(ap, ld_bfrag(sVt + (dt * 16 + n) * 32, 0, lane), o[dt]);
  }

  // normalize and write ctx (bf16, ready as next GEMM A operand)
#pragma unroll
  for (int v = 0; v < 8; ++v) {
    const float inv = 1.0f / lsum[v];
    const size_t row = rowbase + q0 + wave * 16 + v + 8 * g;
#pragma unroll
    for (int dt = 0; dt < 4; ++dt)
      ctx[row * Hc + hoff + dt * 16 + n] = (__bf16)(o[dt][v] * inv);
  }
}

// ---------------------------------------------------------------------------
// Kernel 4: LayerNorm over H=768 cols; writes f32 (residual) + bf16 (GEMM A)
// ---------------------------------------------------------------------------
__device__ __forceinline__ float block_sum768(float v, float* red) {
#pragma unroll
  for (int off = 16; off > 0; off >>= 1) v += __shfl_xor(v, off, 32);
  if ((threadIdx.x & 31) == 0) red[threadIdx.x >> 5] = v;
  __syncthreads();
  float s = (threadIdx.x < 8) ? red[threadIdx.x] : 0.0f;
  s += __shfl_xor(s, 4, 32);
  s += __shfl_xor(s, 2, 32);
  s += __shfl_xor(s, 1, 32);
  if (threadIdx.x == 0) red[0] = s;
  __syncthreads();
  float r = red[0];
  __syncthreads();
  return r;
}

__global__ __launch_bounds__(256) void ln_kernel(const float* __restrict__ in,
                                                 const float* __restrict__ gam,
                                                 const float* __restrict__ bet,
                                                 float* __restrict__ outF,
                                                 __bf16* __restrict__ outB) {
  __shared__ float red[8];
  const size_t row = blockIdx.x;
  const float* x = in + row * Hc;
  float lx[3];
#pragma unroll
  for (int i = 0; i < 3; ++i) lx[i] = x[threadIdx.x + i * 256];
  float u = block_sum768(lx[0] + lx[1] + lx[2], red) * (1.0f / Hc);
  float d0 = lx[0] - u, d1 = lx[1] - u, d2 = lx[2] - u;
  float var = block_sum768(d0 * d0 + d1 * d1 + d2 * d2, red) * (1.0f / Hc);
  float inv = rsqrtf(var + 1e-12f);
#pragma unroll
  for (int i = 0; i < 3; ++i) {
    int c = threadIdx.x + i * 256;
    float y = gam[c] * (lx[i] - u) * inv + bet[c];
    outF[row * Hc + c] = y;
    outB[row * Hc + c] = (__bf16)y;
  }
}

// ---------------------------------------------------------------------------
// Host launcher
// ---------------------------------------------------------------------------
extern "C" void kernel_launch(void* const* d_in, const int* in_sizes, int n_in,
                              void* d_out, int out_size, void* d_ws, size_t ws_size,
                              hipStream_t stream) {
  (void)in_sizes; (void)n_in; (void)out_size; (void)ws_size;
  const float* x     = (const float*)d_in[0];
  const float* graph = (const float*)d_in[1];
  const float* pos   = (const float*)d_in[2];
  const float* in_W  = (const float*)d_in[3];
  const float* in_b  = (const float*)d_in[4];
  const float* emb_g = (const float*)d_in[5];
  const float* emb_b = (const float*)d_in[6];
  const float* Wq    = (const float*)d_in[7];
  const float* Wk    = (const float*)d_in[8];
  const float* Wv    = (const float*)d_in[9];
  const float* so_W  = (const float*)d_in[10];
  const float* so_b  = (const float*)d_in[11];
  const float* so_g  = (const float*)d_in[12];
  const float* so_bt = (const float*)d_in[13];
  const float* i_W   = (const float*)d_in[14];
  const float* i_b   = (const float*)d_in[15];
  const float* o_W   = (const float*)d_in[16];
  const float* o_b   = (const float*)d_in[17];
  const float* o_g   = (const float*)d_in[18];
  const float* o_bt  = (const float*)d_in[19];
  float* out = (float*)d_out;

  char* ws = (char*)d_ws;
  size_t off = 0;
  auto carve = [&](size_t bytes) -> char* {
    char* p = ws + off;
    off = (off + bytes + 255) & ~(size_t)255;
    return p;
  };
  __bf16* xb     = (__bf16*)carve((size_t)Mr * INc * 2);
  float*  pre    = (float*) carve((size_t)Mr * Hc * 4);
  float*  hF     = (float*) carve((size_t)Mr * Hc * 4);
  __bf16* hB     = (__bf16*)carve((size_t)Mr * Hc * 2);
  float*  aF     = (float*) carve((size_t)Mr * Hc * 4);
  __bf16* aB     = (__bf16*)carve((size_t)Mr * Hc * 2);
  __bf16* qB     = (__bf16*)carve((size_t)Mr * Hc * 2);
  __bf16* kB     = (__bf16*)carve((size_t)Mr * Hc * 2);
  __bf16* vB     = (__bf16*)carve((size_t)Mr * Hc * 2);
  __bf16* ctxB   = (__bf16*)carve((size_t)Mr * Hc * 2);
  __bf16* interB = (__bf16*)carve((size_t)Mr * INTc * 2);

  const dim3 blk(256);
  const dim3 gH(Hc / 128, Mr / 128);     // (6, 32)
  const dim3 gI(INTc / 128, Mr / 128);   // (24, 32)

  // Embedding: (x + pos) @ in_W + in_b, then LN
  xpos_kernel<<<(Mr * INc) / 256, blk, 0, stream>>>(x, pos, xb);
  gemm_kernel<true, false, false, true, false><<<gH, blk, 0, stream>>>(
      xb, in_W, Mr, Hc, INc, in_b, nullptr, pre, nullptr);
  ln_kernel<<<Mr, blk, 0, stream>>>(pre, emb_g, emb_b, hF, hB);

  for (int l = 0; l < Lc; ++l) {
    const float* wq = Wq + (size_t)l * Hc * Hc;
    const float* wk = Wk + (size_t)l * Hc * Hc;
    const float* wv = Wv + (size_t)l * Hc * Hc;
    gemm_kernel<false, false, false, false, true><<<gH, blk, 0, stream>>>(
        hB, wq, Mr, Hc, Hc, nullptr, nullptr, nullptr, qB);
    gemm_kernel<false, false, false, false, true><<<gH, blk, 0, stream>>>(
        hB, wk, Mr, Hc, Hc, nullptr, nullptr, nullptr, kB);
    gemm_kernel<false, false, false, false, true><<<gH, blk, 0, stream>>>(
        hB, wv, Mr, Hc, Hc, nullptr, nullptr, nullptr, vB);

    attn_kernel<<<Bc * NHc * (Sc / 128), blk, 0, stream>>>(qB, kB, vB, graph, ctxB);

    // self-output: ctx @ so_W + so_b + h -> LN
    gemm_kernel<true, true, false, true, false><<<gH, blk, 0, stream>>>(
        ctxB, so_W + (size_t)l * Hc * Hc, Mr, Hc, Hc, so_b + (size_t)l * Hc, hF,
        pre, nullptr);
    ln_kernel<<<Mr, blk, 0, stream>>>(pre, so_g + (size_t)l * Hc,
                                      so_bt + (size_t)l * Hc, aF, aB);

    // FFN
    gemm_kernel<true, false, true, false, true><<<gI, blk, 0, stream>>>(
        aB, i_W + (size_t)l * Hc * INTc, Mr, INTc, Hc, i_b + (size_t)l * INTc,
        nullptr, nullptr, interB);
    gemm_kernel<true, true, false, true, false><<<gH, blk, 0, stream>>>(
        interB, o_W + (size_t)l * INTc * Hc, Mr, Hc, INTc, o_b + (size_t)l * Hc,
        aF, pre, nullptr);
    float* dstF = (l == Lc - 1) ? out : hF;
    ln_kernel<<<Mr, blk, 0, stream>>>(pre, o_g + (size_t)l * Hc,
                                      o_bt + (size_t)l * Hc, dstF, hB);
  }
}
